// TransformerRL_52132313038996
// MI455X (gfx1250) — compile-verified, matrix operand order checked by reading, not run on hardware
//
#include <hip/hip_runtime.h>
#include <hip/hip_bf16.h>

// ---------------------------------------------------------------------------
// TransformerRL forward for MI455X (gfx1250): bf16 WMMA everywhere, f32 accum.
// B=4 S=1024 SD=512 D=1024 H=16 L=6 FF=4096 A=64 DK=64
// Double-buffered LDS GEMM pipeline; async global->LDS copies (ASYNCcnt).
// ---------------------------------------------------------------------------

typedef __bf16 bf16;
typedef __attribute__((ext_vector_type(16))) __bf16 v16bf;
typedef __attribute__((ext_vector_type(8)))  __bf16 v8bf;
typedef __attribute__((ext_vector_type(8)))  float  v8f;
typedef __attribute__((ext_vector_type(4)))  int    v4i;

#define TB 4
#define TS 1024
#define TSD 512
#define TD 1024
#define TH 16
#define TL 6
#define TFF 4096
#define TA 64
#define TDK 64
#define TM (TB * TS)   // 4096 rows of activations

#if __has_builtin(__builtin_amdgcn_global_load_async_to_lds_b128)
#define HAVE_ASYNC_LDS 1
#else
#define HAVE_ASYNC_LDS 0
#endif

#define AS1 __attribute__((address_space(1)))
#define AS3 __attribute__((address_space(3)))

// 16-byte global -> LDS copy; async (ASYNCcnt-tracked) when the builtin exists.
static __device__ inline void copy16_g2l(const bf16* g, bf16* l) {
#if HAVE_ASYNC_LDS
    __builtin_amdgcn_global_load_async_to_lds_b128(
        (AS1 v4i*)(unsigned long long)g,
        (AS3 v4i*)(unsigned)(unsigned long long)l, 0, 0);
#else
    *(uint4*)l = *(const uint4*)g;
#endif
}

static __device__ inline void async_wait() {
#if HAVE_ASYNC_LDS
#if __has_builtin(__builtin_amdgcn_s_wait_asynccnt)
    __builtin_amdgcn_s_wait_asynccnt(0);
#else
    asm volatile("s_wait_asynccnt 0" ::: "memory");
#endif
#endif
}

static __device__ inline v8f zero8() {
    v8f z = {0.f, 0.f, 0.f, 0.f, 0.f, 0.f, 0.f, 0.f};
    return z;
}

// Pack two bf16 into one dword for ds_store_b32 transpose writes.
static __device__ inline unsigned pack2(bf16 a, bf16 b) {
    union { bf16 h[2]; unsigned u; } p;
    p.h[0] = a; p.h[1] = b;
    return p.u;
}

// Build a 16x32 bf16 WMMA fragment from LDS (ISA 16-bit A 16x32 layout):
//   lane half lh: elem j (0..7) -> K = lh*8 + j ; elem j+8 -> K = 16 + lh*8 + j
static __device__ inline v16bf load_frag32(const bf16* seg, int lh) {
    v8bf lo = *(const v8bf*)(seg + lh * 8);
    v8bf hi = *(const v8bf*)(seg + 16 + lh * 8);
    v16bf r;
#pragma unroll
    for (int j = 0; j < 8; ++j) { r[j] = lo[j]; r[j + 8] = hi[j]; }
    return r;
}

// ---------------------------------------------------------------------------
// fp32 -> bf16 conversion (4 elems / thread)
// ---------------------------------------------------------------------------
__global__ __launch_bounds__(256) void cvt_bf16_kernel(const float* __restrict__ in,
                                                       bf16* __restrict__ out, int n) {
    int i = (blockIdx.x * 256 + threadIdx.x) * 4;
    if (i + 3 < n) {
        float4 v = *(const float4*)(in + i);
        out[i + 0] = (bf16)v.x;
        out[i + 1] = (bf16)v.y;
        out[i + 2] = (bf16)v.z;
        out[i + 3] = (bf16)v.w;
    } else {
        for (int k = i; k < n; ++k) out[k] = (bf16)in[k];
    }
}

// ---------------------------------------------------------------------------
// Tiled bf16 WMMA GEMM:  C[M,N] = epilogue( A[M,K] @ W[K,N] )
// epilogue: v = (acc + bias[col]) * scale + pe[(row&1023)*N + col]; relu; ->f32/bf16
// BM=BN=128, BK=32; 256 threads = 8 waves; wave tile 32x64 (2x4 WMMA tiles).
// Double-buffered LDS: tile t+1 loads overlap tile t WMMA compute.
// ---------------------------------------------------------------------------
#define BM 128
#define BN 128
#define BK 32
#define LDT 40   // padded LDS stride (bf16); 80B rows keep 16B alignment

__global__ __launch_bounds__(256)
void gemm_bf16_kernel(const bf16* __restrict__ A, const bf16* __restrict__ W,
                      const float* __restrict__ bias, const float* __restrict__ pe,
                      float* __restrict__ outF, bf16* __restrict__ outB,
                      int M, int N, int K, float scale, int relu) {
    __shared__ __align__(16) bf16 As[2][BM][LDT];   // [buf][row][k]
    __shared__ __align__(16) bf16 Bs[2][BN][LDT];   // [buf][n][k] (transposed)

    const int tid  = threadIdx.x;
    const int lane = tid & 31;
    const int wid  = tid >> 5;
    const int l15  = lane & 15;
    const int lh   = lane >> 4;
    const int wave_m = wid & 3;   // 0..3 -> 32-row slabs
    const int wave_n = wid >> 2;  // 0..1 -> 64-col slabs

    const int bm0 = blockIdx.y * BM;
    const int bn0 = blockIdx.x * BN;

    // Loader mappings (256 threads)
    const int a_row = tid >> 1, a_seg = tid & 1;        // 16 bf16 x2 per thread
    const int b_k2  = (tid >> 4) * 2;                   // even k-row pair 0..30
    const int b_n0  = (tid & 15) * 8;                   // 8-col segment

    auto load_tiles = [&](int k0, int buf) {
        // ---- A tile: straight copy (async when available) ----
        const bf16* ga = A + (size_t)(bm0 + a_row) * K + k0 + a_seg * 16;
        bf16* la = &As[buf][a_row][a_seg * 16];
        copy16_g2l(ga, la);
        copy16_g2l(ga + 8, la + 8);
        // ---- W tile: 2 k-rows x 8 cols, packed b32 transpose stores ----
        const bf16* gw = W + (size_t)(k0 + b_k2) * N + bn0 + b_n0;
        v8bf r0 = *(const v8bf*)gw;
        v8bf r1 = *(const v8bf*)(gw + N);
#pragma unroll
        for (int j = 0; j < 8; ++j)
            *(unsigned*)&Bs[buf][b_n0 + j][b_k2] = pack2(r0[j], r1[j]);
        // prefetch tile after next into caches
        if (k0 + BK < K) {
            __builtin_prefetch(ga + BK, 0, 3);
            __builtin_prefetch(gw + (size_t)BK * N, 0, 3);
        }
    };

    v8f acc[2][4];
#pragma unroll
    for (int i = 0; i < 2; ++i)
#pragma unroll
        for (int j = 0; j < 4; ++j) acc[i][j] = zero8();

    const int nk = K / BK;
    load_tiles(0, 0);
    async_wait();
    __syncthreads();

    for (int t = 0; t < nk; ++t) {
        const int cur = t & 1;
        if (t + 1 < nk) load_tiles((t + 1) * BK, cur ^ 1);

        v16bf afrag[2], bfrag[4];
#pragma unroll
        for (int i = 0; i < 2; ++i)
            afrag[i] = load_frag32(&As[cur][wave_m * 32 + i * 16 + l15][0], lh);
#pragma unroll
        for (int j = 0; j < 4; ++j)
            bfrag[j] = load_frag32(&Bs[cur][wave_n * 64 + j * 16 + l15][0], lh);

#pragma unroll
        for (int i = 0; i < 2; ++i)
#pragma unroll
            for (int j = 0; j < 4; ++j)
                acc[i][j] = __builtin_amdgcn_wmma_f32_16x16x32_bf16(
                    false, afrag[i], false, bfrag[j], (short)0, acc[i][j], false, false);

        async_wait();
        __syncthreads();
    }

    // ---- epilogue (C layout: vgpr r, lane -> row r + lh*8, col l15) ----
#pragma unroll
    for (int i = 0; i < 2; ++i) {
#pragma unroll
        for (int j = 0; j < 4; ++j) {
            const int col = bn0 + wave_n * 64 + j * 16 + l15;
            float bcol = bias ? bias[col] : 0.f;
#pragma unroll
            for (int r = 0; r < 8; ++r) {
                const int row = bm0 + wave_m * 32 + i * 16 + r + lh * 8;
                float v = (acc[i][j][r] + bcol) * scale;
                if (pe) v += pe[(size_t)(row & (TS - 1)) * N + col];
                if (relu) v = fmaxf(v, 0.f);
                const size_t idx = (size_t)row * N + col;
                if (outF) outF[idx] = v;
                if (outB) outB[idx] = (bf16)v;
            }
        }
    }
}

// ---------------------------------------------------------------------------
// Flash attention: block = 128 threads (4 waves), one (b, h, 64-query tile).
// q is pre-scaled by 1/sqrt(DK). Online softmax, O accumulated in f32.
// ---------------------------------------------------------------------------
__global__ __launch_bounds__(128)
void attn_kernel(const bf16* __restrict__ q, const bf16* __restrict__ k,
                 const bf16* __restrict__ v, bf16* __restrict__ ctx) {
    __shared__ __align__(16) bf16 Qs[64][72];   // [qrow][dk]
    __shared__ __align__(16) bf16 Ks[64][72];   // [key][dk]
    __shared__ __align__(16) bf16 Vs[64][72];   // transposed: [dk][key]
    __shared__ __align__(16) bf16 Ps[64][72];   // [qrow][key]

    const int qt = blockIdx.x, h = blockIdx.y, b = blockIdx.z;
    const int tid = threadIdx.x;
    const int lane = tid & 31, w = tid >> 5;
    const int l15 = lane & 15, lh = lane >> 4;

    const int ldr = tid >> 1, lds2 = (tid & 1) * 32;   // Q/K loader: row, 32-elem seg
    const int kp  = (tid >> 2) * 2;                    // V loader: even key pair
    const int dk0 = (tid & 3) * 16;                    // V loader: 16-dk segment

    // ---- load Q tile (async copies) ----
    {
        const bf16* g = q + ((size_t)(b * TS + qt * 64 + ldr)) * TD + h * TDK + lds2;
        bf16* d = &Qs[ldr][lds2];
#pragma unroll
        for (int c = 0; c < 4; ++c) copy16_g2l(g + c * 8, d + c * 8);
    }

    float mrow[8], srow[8];
    v8f o[4];
#pragma unroll
    for (int r = 0; r < 8; ++r) { mrow[r] = -1e30f; srow[r] = 0.f; }
#pragma unroll
    for (int t = 0; t < 4; ++t) o[t] = zero8();

    for (int jt = 0; jt < TS / 64; ++jt) {
        __syncthreads();   // previous P@V finished reading Ks/Vs/Ps
        // ---- load K tile (async copies) ----
        {
            const bf16* g = k + ((size_t)(b * TS + jt * 64 + ldr)) * TD + h * TDK + lds2;
            bf16* d = &Ks[ldr][lds2];
#pragma unroll
            for (int c = 0; c < 4; ++c) copy16_g2l(g + c * 8, d + c * 8);
        }
        // ---- load V tile: 2 keys x 16 dk, packed b32 transpose stores ----
        {
            const bf16* g0 = v + ((size_t)(b * TS + jt * 64 + kp)) * TD + h * TDK + dk0;
            v8bf a0 = *(const v8bf*)g0,        a1 = *(const v8bf*)(g0 + 8);
            v8bf c0 = *(const v8bf*)(g0 + TD), c1 = *(const v8bf*)(g0 + TD + 8);
#pragma unroll
            for (int j = 0; j < 8; ++j) {
                *(unsigned*)&Vs[dk0 + j][kp]     = pack2(a0[j], c0[j]);
                *(unsigned*)&Vs[dk0 + 8 + j][kp] = pack2(a1[j], c1[j]);
            }
        }
        async_wait();
        __syncthreads();

        // ---- scores S = Q @ K^T (wave w owns q rows w*16..w*16+15) ----
        v8f sc[4];
#pragma unroll
        for (int kt = 0; kt < 4; ++kt) sc[kt] = zero8();
#pragma unroll
        for (int kk = 0; kk < 2; ++kk) {
            v16bf aq = load_frag32(&Qs[w * 16 + l15][kk * 32], lh);
#pragma unroll
            for (int kt = 0; kt < 4; ++kt) {
                v16bf bk = load_frag32(&Ks[kt * 16 + l15][kk * 32], lh);
                sc[kt] = __builtin_amdgcn_wmma_f32_16x16x32_bf16(
                    false, aq, false, bk, (short)0, sc[kt], false, false);
            }
        }

        // ---- online softmax (row = r + lh*8; 16 lanes/row share state) ----
#pragma unroll
        for (int r = 0; r < 8; ++r) {
            float tm = sc[0][r];
#pragma unroll
            for (int kt = 1; kt < 4; ++kt) tm = fmaxf(tm, sc[kt][r]);
#pragma unroll
            for (int msk = 1; msk < 16; msk <<= 1) tm = fmaxf(tm, __shfl_xor(tm, msk, 32));
            const float nm = fmaxf(mrow[r], tm);
            const float alpha = __expf(mrow[r] - nm);
            mrow[r] = nm;
            float ps = 0.f;
#pragma unroll
            for (int kt = 0; kt < 4; ++kt) {
                float p = __expf(sc[kt][r] - nm);
                sc[kt][r] = p;
                ps += p;
            }
#pragma unroll
            for (int msk = 1; msk < 16; msk <<= 1) ps += __shfl_xor(ps, msk, 32);
            srow[r] = srow[r] * alpha + ps;
#pragma unroll
            for (int t = 0; t < 4; ++t) o[t][r] *= alpha;
        }

        // ---- spill P to LDS in A-fragment layout ----
#pragma unroll
        for (int kt = 0; kt < 4; ++kt)
#pragma unroll
            for (int r = 0; r < 8; ++r)
                Ps[w * 16 + r + lh * 8][kt * 16 + l15] = (bf16)sc[kt][r];
        __syncthreads();

        // ---- O += P @ V ----
#pragma unroll
        for (int kk = 0; kk < 2; ++kk) {
            v16bf ap = load_frag32(&Ps[w * 16 + l15][kk * 32], lh);
#pragma unroll
            for (int t = 0; t < 4; ++t) {
                v16bf bv = load_frag32(&Vs[t * 16 + l15][kk * 32], lh);
                o[t] = __builtin_amdgcn_wmma_f32_16x16x32_bf16(
                    false, ap, false, bv, (short)0, o[t], false, false);
            }
        }
    }

    // ---- normalize + write ctx (B*S, D) with head offset ----
#pragma unroll
    for (int r = 0; r < 8; ++r) {
        const float inv = 1.0f / srow[r];
        const int row = b * TS + qt * 64 + w * 16 + r + lh * 8;
#pragma unroll
        for (int t = 0; t < 4; ++t) {
            const int col = h * TDK + t * 16 + l15;
            ctx[(size_t)row * TD + col] = (bf16)(o[t][r] * inv);
        }
    }
}

// ---------------------------------------------------------------------------
// x = LayerNorm(x + y) * s + b ; writes f32 and bf16 copies. One row / block.
// ---------------------------------------------------------------------------
__global__ __launch_bounds__(256)
void add_ln_kernel(const float* __restrict__ x, const float* __restrict__ y,
                   const float* __restrict__ s, const float* __restrict__ bta,
                   float* __restrict__ xo, bf16* __restrict__ xbo) {
    const int row = blockIdx.x;
    const int tid = threadIdx.x;
    const int lane = tid & 31, w = tid >> 5;

    const float4 xv = ((const float4*)(x + (size_t)row * TD))[tid];
    const float4 yv = ((const float4*)(y + (size_t)row * TD))[tid];
    float t[4] = {xv.x + yv.x, xv.y + yv.y, xv.z + yv.z, xv.w + yv.w};

    float sum = t[0] + t[1] + t[2] + t[3];
    float sq  = t[0] * t[0] + t[1] * t[1] + t[2] * t[2] + t[3] * t[3];
#pragma unroll
    for (int msk = 16; msk; msk >>= 1) {
        sum += __shfl_xor(sum, msk, 32);
        sq  += __shfl_xor(sq,  msk, 32);
    }
    __shared__ float s1[8], s2[8];
    if (!lane) { s1[w] = sum; s2[w] = sq; }
    __syncthreads();
    float S = 0.f, Q = 0.f;
#pragma unroll
    for (int i = 0; i < 8; ++i) { S += s1[i]; Q += s2[i]; }
    const float mu = S * (1.0f / TD);
    const float var = Q * (1.0f / TD) - mu * mu;
    const float rs = rsqrtf(var + 1e-5f);
#pragma unroll
    for (int i = 0; i < 4; ++i) {
        const int col = tid * 4 + i;
        const float vv = (t[i] - mu) * rs * s[col] + bta[col];
        xo[(size_t)row * TD + col] = vv;
        xbo[(size_t)row * TD + col] = (bf16)vv;
    }
}

// ---------------------------------------------------------------------------
// Heads: policy (B,64) then value (B,1), concatenated into d_out.
// ---------------------------------------------------------------------------
__global__ __launch_bounds__(64)
void heads_kernel(const float* __restrict__ x, const float* __restrict__ phw,
                  const float* __restrict__ phb, const float* __restrict__ vhw,
                  const float* __restrict__ vhb, float* __restrict__ out) {
    const int b = blockIdx.x;
    const int a = threadIdx.x;
    const float* last = x + ((size_t)b * TS + (TS - 1)) * TD;
    float acc = phb[a];
    for (int d = 0; d < TD; ++d) acc += last[d] * phw[d * TA + a];
    out[b * TA + a] = acc;
    if (a == 0) {
        float vv = vhb[0];
        for (int d = 0; d < TD; ++d) vv += last[d] * vhw[d];
        out[TB * TA + b] = vv;
    }
}

// ---------------------------------------------------------------------------
// Host orchestration
// ---------------------------------------------------------------------------
extern "C" void kernel_launch(void* const* d_in, const int* in_sizes, int n_in,
                              void* d_out, int out_size, void* d_ws, size_t ws_size,
                              hipStream_t stream) {
    (void)in_sizes; (void)n_in; (void)out_size; (void)ws_size;
    const float* states = (const float*)d_in[0];
    const float* pe     = (const float*)d_in[1];
    const float* emb_w  = (const float*)d_in[2];
    const float* emb_b  = (const float*)d_in[3];
    const float* Wq     = (const float*)d_in[4];
    const float* bq     = (const float*)d_in[5];
    const float* Wk     = (const float*)d_in[6];
    const float* bk     = (const float*)d_in[7];
    const float* Wv     = (const float*)d_in[8];
    const float* bv     = (const float*)d_in[9];
    const float* Wo     = (const float*)d_in[10];
    const float* bo     = (const float*)d_in[11];
    const float* ln1_s  = (const float*)d_in[12];
    const float* ln1_b  = (const float*)d_in[13];
    const float* ln2_s  = (const float*)d_in[14];
    const float* ln2_b  = (const float*)d_in[15];
    const float* ff_w1  = (const float*)d_in[16];
    const float* ff_b1  = (const float*)d_in[17];
    const float* ff_w2  = (const float*)d_in[18];
    const float* ff_b2  = (const float*)d_in[19];
    const float* vh_w   = (const float*)d_in[20];
    const float* vh_b   = (const float*)d_in[21];
    const float* ph_w   = (const float*)d_in[22];
    const float* ph_b   = (const float*)d_in[23];

    char* wsb = (char*)d_ws;
    size_t off = 0;
    auto take = [&](size_t bytes) -> void* {
        void* p = wsb + off;
        off = (off + bytes + 255) & ~(size_t)255;
        return p;
    };

    bf16* states_bf = (bf16*)take((size_t)TB * TS * TSD * 2);
    bf16* embw_bf   = (bf16*)take((size_t)TSD * TD * 2);
    bf16* wq_bf     = (bf16*)take((size_t)TL * TD * TD * 2);
    bf16* wk_bf     = (bf16*)take((size_t)TL * TD * TD * 2);
    bf16* wv_bf     = (bf16*)take((size_t)TL * TD * TD * 2);
    bf16* wo_bf     = (bf16*)take((size_t)TL * TD * TD * 2);
    bf16* f1_bf     = (bf16*)take((size_t)TL * TD * TFF * 2);
    bf16* f2_bf     = (bf16*)take((size_t)TL * TFF * TD * 2);
    float* x_f      = (float*)take((size_t)TM * TD * 4);
    float* y_f      = (float*)take((size_t)TM * TD * 4);
    bf16* x_bf      = (bf16*)take((size_t)TM * TD * 2);
    bf16* q_bf      = (bf16*)take((size_t)TM * TD * 2);
    bf16* k_bf      = (bf16*)take((size_t)TM * TD * 2);
    bf16* v_bf      = (bf16*)take((size_t)TM * TD * 2);
    bf16* ctx_bf    = (bf16*)take((size_t)TM * TD * 2);
    bf16* h_bf      = (bf16*)take((size_t)TM * TFF * 2);

    auto cvt = [&](const float* src, bf16* dst, size_t n) {
        int blocks = (int)((n / 4 + 255) / 256);
        cvt_bf16_kernel<<<blocks, 256, 0, stream>>>(src, dst, (int)n);
    };
    auto gemm = [&](const bf16* A, const bf16* W, const float* bias, const float* pep,
                    float* oF, bf16* oB, int M, int N, int K, float scale, int relu) {
        dim3 g(N / BN, M / BM);
        gemm_bf16_kernel<<<g, 256, 0, stream>>>(A, W, bias, pep, oF, oB, M, N, K, scale, relu);
    };

    // ---- fp32 -> bf16 conversions ----
    cvt(states, states_bf, (size_t)TB * TS * TSD);
    cvt(emb_w, embw_bf, (size_t)TSD * TD);
    cvt(Wq, wq_bf, (size_t)TL * TD * TD);
    cvt(Wk, wk_bf, (size_t)TL * TD * TD);
    cvt(Wv, wv_bf, (size_t)TL * TD * TD);
    cvt(Wo, wo_bf, (size_t)TL * TD * TD);
    cvt(ff_w1, f1_bf, (size_t)TL * TD * TFF);
    cvt(ff_w2, f2_bf, (size_t)TL * TFF * TD);

    // ---- embedding: x = states @ emb_w + emb_b + pe ----
    gemm(states_bf, embw_bf, emb_b, pe, x_f, x_bf, TM, TD, TSD, 1.0f, 0);

    const float qscale = 0.125f;  // 1/sqrt(DK) folded into q
    for (int l = 0; l < TL; ++l) {
        const size_t wofs = (size_t)l * TD * TD;
        gemm(x_bf, wq_bf + wofs, bq + l * TD, nullptr, nullptr, q_bf, TM, TD, TD, qscale, 0);
        gemm(x_bf, wk_bf + wofs, bk + l * TD, nullptr, nullptr, k_bf, TM, TD, TD, 1.0f, 0);
        gemm(x_bf, wv_bf + wofs, bv + l * TD, nullptr, nullptr, v_bf, TM, TD, TD, 1.0f, 0);

        attn_kernel<<<dim3(TS / 64, TH, TB), 128, 0, stream>>>(q_bf, k_bf, v_bf, ctx_bf);

        gemm(ctx_bf, wo_bf + wofs, bo + l * TD, nullptr, y_f, nullptr, TM, TD, TD, 1.0f, 0);
        add_ln_kernel<<<TM, 256, 0, stream>>>(x_f, y_f, ln1_s + l * TD, ln1_b + l * TD, x_f, x_bf);

        gemm(x_bf, f1_bf + (size_t)l * TD * TFF, ff_b1 + l * TFF, nullptr, nullptr, h_bf,
             TM, TFF, TD, 1.0f, 1);
        gemm(h_bf, f2_bf + (size_t)l * TFF * TD, ff_b2 + l * TD, nullptr, y_f, nullptr,
             TM, TD, TFF, 1.0f, 0);
        add_ln_kernel<<<TM, 256, 0, stream>>>(x_f, y_f, ln2_s + l * TD, ln2_b + l * TD, x_f, x_bf);
    }

    heads_kernel<<<TB, 64, 0, stream>>>(x_f, ph_w, ph_b, vh_w, vh_b, (float*)d_out);
}
